// PEM_CA_10788957847662
// MI455X (gfx1250) — compile-verified, matrix operand order checked by reading, not run on hardware
//
#include <hip/hip_runtime.h>
#include <hip/hip_bf16.h>

#define BDIM_B 2
#define QDIM 100
#define NDIM 16384
#define CDIM 256
#define HDIM 8
#define DDIM 32
#define QPAD 112          // 7 tiles of 16
#define BN_EPS 1e-5f
#define LN_EPS 1e-5f

typedef __attribute__((ext_vector_type(16))) __bf16 v16bf;
typedef __attribute__((ext_vector_type(8)))  __bf16 v8bf;
typedef __attribute__((ext_vector_type(8)))  float  v8f;

__device__ __forceinline__ __bf16 f2bf(float f) {
    union { float f; unsigned u; } v; v.f = f;
    unsigned r = v.u + 0x7FFFu + ((v.u >> 16) & 1u);   // round-to-nearest-even
    union { unsigned short s; __bf16 b; } o;
    o.s = (unsigned short)(r >> 16);
    return o.b;
}

// Build a v16bf fragment from two contiguous 8-element (16B) runs.
// Per CDNA5 ISA 16-bit A/B layout: lane (half*16+i) holds K in
// [half*8, half*8+8) (elements 0..7) and [16+half*8, ...) (elements 8..15).
__device__ __forceinline__ v16bf load_frag(const __bf16* base) {
    v8bf lo = *(const v8bf*)(base);
    v8bf hi = *(const v8bf*)(base + 16);
    v16bf r;
#pragma unroll
    for (int i = 0; i < 8; ++i) { r[i] = lo[i]; r[i + 8] = hi[i]; }
    return r;
}

// CDNA5 async global->LDS copy (ASYNCcnt-tracked; no VGPR round-trip).
// lds_off = low 32 bits of the flat shared-memory address (aperture maps addr[31:0] to LDS).
__device__ __forceinline__ void async_copy_b128(unsigned lds_off, const void* gaddr) {
    asm volatile("global_load_async_to_lds_b128 %0, %1, off"
                 :: "v"(lds_off), "v"(gaddr) : "memory");
}
__device__ __forceinline__ void wait_async0() {
    asm volatile("s_wait_asynccnt 0" ::: "memory");
}

__device__ __forceinline__ float blockSum(float v, float* tmp) {
    int t = threadIdx.x;
    tmp[t] = v; __syncthreads();
    for (int s = 128; s > 0; s >>= 1) {
        if (t < s) tmp[t] += tmp[t + s];
        __syncthreads();
    }
    float r = tmp[0]; __syncthreads();
    return r;
}

// ---------------- init: zero argmax table + mask flags ----------------
__global__ void k_init(unsigned long long* argb, int* flag) {
    int i = blockIdx.x * 256 + threadIdx.x;
    if (i < BDIM_B * HDIM * QPAD) argb[i] = 0ull;
    if (i < BDIM_B) flag[i] = 0;
}

// --------- any (mask >= 0) per batch -> decides whether mask applies --------
__global__ void k_anymask(const float* __restrict__ mask, int* flag) {
    int i = blockIdx.x * 256 + threadIdx.x;            // over B*Q*N
    if (i >= BDIM_B * QDIM * NDIM) return;
    int b = i / (QDIM * NDIM);
    if (mask[i] >= 0.f) atomicOr(&flag[b], 1);
}

// ---- t path: (tgt+query_pos) -> LayerNorm -> qW linear -> l2norm(C) ----
__global__ void k_tpath(const float* __restrict__ tgt, const float* __restrict__ qpos,
                        const float* __restrict__ lnW, const float* __restrict__ lnb,
                        const float* __restrict__ qW,  const float* __restrict__ qb,
                        float* __restrict__ tn, __bf16* __restrict__ tbf) {
    __shared__ float sh[CDIM];
    __shared__ float tmp[CDIM];
    int bq = blockIdx.x;            // 0..B*Q-1
    int c  = threadIdx.x;
    float v  = tgt[bq * CDIM + c] + qpos[bq * CDIM + c];
    float mu = blockSum(v, tmp) * (1.f / CDIM);
    float d  = v - mu;
    float var = blockSum(d * d, tmp) * (1.f / CDIM);
    float tln = d * rsqrtf(var + LN_EPS) * lnW[c] + lnb[c];
    sh[c] = tln; __syncthreads();
    float acc = qb[c];
    const float* wr = qW + c * CDIM;
#pragma unroll 8
    for (int k = 0; k < CDIM; ++k) acc += sh[k] * wr[k];
    float nsq = blockSum(acc * acc, tmp);
    float o = acc / fmaxf(sqrtf(nsq), 1e-12f);
    tn[bq * CDIM + c]  = o;
    tbf[bq * CDIM + c] = f2bf(o);
}

// ---- mem path pre-GEMM: (memory+pos) -> BN(inference) -> depthwise conv k=3 -> bf16 ----
__global__ void k_bnconv(const float* __restrict__ memory, const float* __restrict__ pos,
                         const float* __restrict__ gam, const float* __restrict__ bta,
                         const float* __restrict__ mean, const float* __restrict__ var,
                         const float* __restrict__ cw, const float* __restrict__ cb,
                         __bf16* __restrict__ y) {
    int i = blockIdx.x * 256 + threadIdx.x;            // over B*N*C
    int c  = i & (CDIM - 1);
    int bn = i >> 8;
    int n  = bn & (NDIM - 1);
    int b  = bn >> 14;
    float sc = gam[c] * rsqrtf(var[c] + BN_EPS);
    float shv = bta[c] - mean[c] * sc;
    float w0 = cw[c * 3 + 0], w1 = cw[c * 3 + 1], w2 = cw[c * 3 + 2];
    float acc = cb[c];
    size_t base = ((size_t)b * NDIM) * CDIM + c;
    if (n > 0) {
        size_t j = base + (size_t)(n - 1) * CDIM;
        acc += ((memory[j] + pos[j]) * sc + shv) * w0;
    }
    {
        size_t j = base + (size_t)n * CDIM;
        acc += ((memory[j] + pos[j]) * sc + shv) * w1;
    }
    if (n < NDIM - 1) {
        size_t j = base + (size_t)(n + 1) * CDIM;
        acc += ((memory[j] + pos[j]) * sc + shv) * w2;
    }
    y[i] = f2bf(acc);
}

__global__ void k_w2bf(const float* __restrict__ w, __bf16* __restrict__ o) {
    int i = blockIdx.x * 256 + threadIdx.x;            // C*C
    o[i] = f2bf(w[i]);
}

// ---- big GEMM: out[bn,o] = sum_c y[bn,c]*featW[o,c] + featb[o]  (WMMA bf16) ----
// grid (M/16, 2), block 128 (4 waves).
// featW slice (16 O-rows x 256 K = 8 KB bf16) is staged ONCE per block into LDS with
// CDNA5 async global->LDS b128 copies (ASYNCcnt), then the steady-state loop streams only
// A from global (read exactly once = roofline minimum) while B fragments come from LDS.
__global__ void k_gemm_feat(const __bf16* __restrict__ A, const __bf16* __restrict__ Wb,
                            const float* __restrict__ bias, float* __restrict__ out) {
    __shared__ __bf16 shW[128 * CDIM];                 // 64 KB of the 320 KB WGP pool
    int tid  = threadIdx.x;                            // 0..127
    int lane = tid & 31;
    int wave = tid >> 5;                               // 0..3
    int mBase = blockIdx.x * 16;
    int oBlk  = blockIdx.y * 128;

    // ---- async stage featW[oBlk .. oBlk+16) x K into LDS: 512 x b128, 4 per lane-slot ----
    {
        unsigned lbase = (unsigned)(unsigned long long)(&shW[0]);
        const __bf16* gbase = Wb + (size_t)oBlk * CDIM;
#pragma unroll
        for (int j = 0; j < 4; ++j) {
            int chunk = tid + 128 * j;                 // each chunk = 8 bf16 = 16 bytes
            async_copy_b128(lbase + chunk * 16, (const void*)(gbase + chunk * 8));
        }
    }
    wait_async0();                                     // each wave drains its own ASYNCcnt...
    __syncthreads();                                   // ...then barrier makes LDS tile visible to all

    int half = lane >> 4, l15 = lane & 15;
    int oLoc = wave * 32;                              // this wave's 32 O-rows within the LDS tile
    const __bf16* aRow  = A   + (size_t)(mBase + l15) * CDIM + half * 8;
    const __bf16* b0Row = shW + (size_t)(oLoc + l15) * CDIM + half * 8;
    const __bf16* b1Row = shW + (size_t)(oLoc + 16 + l15) * CDIM + half * 8;
    v8f acc0 = {}; v8f acc1 = {};
#pragma unroll
    for (int k = 0; k < CDIM; k += 32) {
        v16bf a  = load_frag(aRow + k);                // global (streamed once)
        v16bf b0 = load_frag(b0Row + k);               // LDS -> ds_load_b128
        v16bf b1 = load_frag(b1Row + k);
        acc0 = __builtin_amdgcn_wmma_f32_16x16x32_bf16(false, a, false, b0, (short)0, acc0, false, false);
        acc1 = __builtin_amdgcn_wmma_f32_16x16x32_bf16(false, a, false, b1, (short)0, acc1, false, false);
    }
    int oBase = oBlk + oLoc;
#pragma unroll
    for (int r = 0; r < 8; ++r) {
        int row = mBase + half * 8 + r;
        out[(size_t)row * CDIM + oBase + l15]      = acc0[r] + bias[oBase + l15];
        out[(size_t)row * CDIM + oBase + 16 + l15] = acc1[r] + bias[oBase + 16 + l15];
    }
}

// ---- l2-normalize each projected mem row (axis=C); keep f32 (for gather) + bf16 (for WMMA sim) ----
__global__ void k_rownorm(float* __restrict__ mp, __bf16* __restrict__ mbf) {
    __shared__ float tmp[CDIM];
    size_t row = blockIdx.x;
    int c = threadIdx.x;
    float v = mp[row * CDIM + c];
    float nsq = blockSum(v * v, tmp);
    float o = v / fmaxf(sqrtf(nsq), 1e-12f);
    mp[row * CDIM + c]  = o;
    mbf[row * CDIM + c] = f2bf(o);
}

// ---- fused similarity (WMMA, K=32=head dim) + masked argmax via packed u64 atomicMax ----
// grid (N/128, H, B), block 256 (8 waves x 16 keys each). Never materializes sim[B,H,N,Q].
__global__ void k_sim(const __bf16* __restrict__ keys, const __bf16* __restrict__ qs,
                      const float* __restrict__ mask, const int* __restrict__ flag,
                      unsigned long long* __restrict__ argb) {
    int b = blockIdx.z, h = blockIdx.y;
    int lane = threadIdx.x & 31, wave = threadIdx.x >> 5;
    int nBase = blockIdx.x * 128 + wave * 16;
    int half = lane >> 4, l15 = lane & 15;
    bool use_mask = (flag[b] != 0);
    const __bf16* aRow = keys + ((size_t)(b * NDIM + nBase + l15)) * CDIM + h * DDIM + half * 8;
    v16bf a = load_frag(aRow);                         // 16 keys x K=32 (= D), exactly one WMMA step
    for (int qt = 0; qt < QPAD / 16; ++qt) {
        int q  = qt * 16 + l15;
        int qc = q < QDIM ? q : QDIM - 1;
        const __bf16* bRow = qs + ((size_t)(b * QDIM + qc)) * CDIM + h * DDIM + half * 8;
        v16bf bb = load_frag(bRow);
        v8f c = {};
        c = __builtin_amdgcn_wmma_f32_16x16x32_bf16(false, a, false, bb, (short)0, c, false, false);
        // this lane: one query column, 8 key rows (n = nBase + half*8 + r); mask row is contiguous in n
        float best = -__builtin_inff();
        int   bestn = nBase + half * 8;
        const float* mrow = mask + ((size_t)(b * QDIM + qc)) * NDIM + nBase + half * 8;
#pragma unroll
        for (int r = 0; r < 8; ++r) {
            float v = c[r];
            if (use_mask && mrow[r] < 0.f) v = -__builtin_inff();
            int nn = nBase + half * 8 + r;
            if (v > best) { best = v; bestn = nn; }
        }
        // merge the two 8-row halves (lane L <-> L+16 hold the same query)
        float ov = __shfl_xor(best, 16);
        int   oi = __shfl_xor(bestn, 16);
        if (ov > best || (ov == best && oi < bestn)) { best = ov; bestn = oi; }
        if (half == 0 && q < QDIM) {
            unsigned u = __float_as_uint(best);
            unsigned key = (u & 0x80000000u) ? ~u : (u | 0x80000000u);  // order-preserving f32->u32
            // inverted index so ties prefer the LOWER n under atomicMax (argmax-first semantics)
            unsigned long long packed =
                ((unsigned long long)key << 32) | (unsigned long long)(0xFFFFFFFFu - (unsigned)bestn);
            atomicMax(&argb[(size_t)(b * HDIM + h) * QPAD + q], packed);
        }
    }
}

// ---- gather winning keys per (b,q,head) + elementwise mem*t ----
__global__ void k_gather(const float* __restrict__ memn, const float* __restrict__ tn,
                         const unsigned long long* __restrict__ argb,
                         float* __restrict__ g, float* __restrict__ gm) {
    int i = blockIdx.x * 256 + threadIdx.x;            // over B*Q*C
    int c = i & (CDIM - 1);
    int bq = i >> 8;
    int q = bq % QDIM, b = bq / QDIM;
    int h = c >> 5;
    unsigned long long p = argb[(size_t)(b * HDIM + h) * QPAD + q];
    unsigned n = 0xFFFFFFFFu - (unsigned)(p & 0xFFFFFFFFu);
    if (n >= NDIM) n = 0;
    float gv = memn[((size_t)(b * NDIM + n)) * CDIM + c];
    g[i]  = gv;
    gm[i] = gv * tn[i];
}

// ---- small linear: out[bq,o] = sum_c x[bq,c]*W[o,c] + bias[o] ----
__global__ void k_linear(const float* __restrict__ x, const float* __restrict__ W,
                         const float* __restrict__ bias, float* __restrict__ out) {
    __shared__ float sh[CDIM];
    int bq = blockIdx.x, o = threadIdx.x;
    sh[o] = x[bq * CDIM + o]; __syncthreads();
    float acc = bias[o];
    const float* wr = W + o * CDIM;
#pragma unroll 8
    for (int k = 0; k < CDIM; ++k) acc += sh[k] * wr[k];
    out[bq * CDIM + o] = acc;
}

// ---- column norms over Q (l2norm axis=1) ----
__global__ void k_colnorm(const float* __restrict__ p, float* __restrict__ norm) {
    int b = blockIdx.x, c = threadIdx.x;
    float acc = 0.f;
    for (int q = 0; q < QDIM; ++q) {
        float v = p[((size_t)(b * QDIM + q)) * CDIM + c];
        acc += v * v;
    }
    norm[b * CDIM + c] = fmaxf(sqrtf(acc), 1e-12f);
}

// ---- finish: out2 = p/norm*alpha + g ; out = out2 @ finW^T + finb + tgt ----
__global__ void k_finish(const float* __restrict__ p, const float* __restrict__ norm,
                         const float* __restrict__ alpha, const float* __restrict__ g,
                         const float* __restrict__ W, const float* __restrict__ bias,
                         const float* __restrict__ tgt, float* __restrict__ out) {
    __shared__ float sh[CDIM];
    int bq = blockIdx.x, b = bq / QDIM, o = threadIdx.x;
    float o2 = p[bq * CDIM + o] / norm[b * CDIM + o] * alpha[o] + g[bq * CDIM + o];
    sh[o] = o2; __syncthreads();
    float acc = bias[o];
    const float* wr = W + o * CDIM;
#pragma unroll 8
    for (int k = 0; k < CDIM; ++k) acc += sh[k] * wr[k];
    out[bq * CDIM + o] = acc + tgt[bq * CDIM + o];
}

extern "C" void kernel_launch(void* const* d_in, const int* in_sizes, int n_in,
                              void* d_out, int out_size, void* d_ws, size_t ws_size,
                              hipStream_t stream) {
    const float* tgt      = (const float*)d_in[0];
    const float* memory   = (const float*)d_in[1];
    const float* mmask    = (const float*)d_in[2];
    const float* pos      = (const float*)d_in[3];
    const float* qpos     = (const float*)d_in[4];
    const float* bn_gamma = (const float*)d_in[5];
    const float* bn_beta  = (const float*)d_in[6];
    const float* bn_mean  = (const float*)d_in[7];
    const float* bn_var   = (const float*)d_in[8];
    const float* conv_w   = (const float*)d_in[9];
    const float* conv_b   = (const float*)d_in[10];
    const float* featW    = (const float*)d_in[11];
    const float* featb    = (const float*)d_in[12];
    const float* lnW      = (const float*)d_in[13];
    const float* lnb      = (const float*)d_in[14];
    const float* qW       = (const float*)d_in[15];
    const float* qb       = (const float*)d_in[16];
    const float* projW    = (const float*)d_in[17];
    const float* projb    = (const float*)d_in[18];
    const float* finW     = (const float*)d_in[19];
    const float* finb     = (const float*)d_in[20];
    const float* alpha    = (const float*)d_in[21];
    float* out = (float*)d_out;

    char* w = (char*)d_ws;
    auto alloc = [&](size_t bytes) -> void* {
        void* p = (void*)w;
        w += (bytes + 255) & ~(size_t)255;
        return p;
    };
    const size_t BN  = (size_t)BDIM_B * NDIM;          // 32768
    const size_t BQ  = (size_t)BDIM_B * QDIM;          // 200
    __bf16* ybf   = (__bf16*)alloc(BN * CDIM * 2);     // conv output bf16; reused as normalized mem bf16
    __bf16* wbf   = (__bf16*)alloc((size_t)CDIM * CDIM * 2);
    float*  mproj = (float*) alloc(BN * CDIM * 4);     // feature GEMM out, normalized in place
    float*  tn    = (float*) alloc(BQ * CDIM * 4);
    __bf16* tbf   = (__bf16*)alloc(BQ * CDIM * 2);
    unsigned long long* argb = (unsigned long long*)alloc((size_t)BDIM_B * HDIM * QPAD * 8);
    int*    flag  = (int*)   alloc(64);
    float*  g     = (float*) alloc(BQ * CDIM * 4);
    float*  gm    = (float*) alloc(BQ * CDIM * 4);
    float*  pout  = (float*) alloc(BQ * CDIM * 4);
    float*  nrm   = (float*) alloc((size_t)BDIM_B * CDIM * 4);

    k_init<<<7, 256, 0, stream>>>(argb, flag);
    k_anymask<<<(BDIM_B * QDIM * NDIM) / 256, 256, 0, stream>>>(mmask, flag);
    k_tpath<<<(int)BQ, 256, 0, stream>>>(tgt, qpos, lnW, lnb, qW, qb, tn, tbf);
    k_bnconv<<<(int)(BN * CDIM / 256), 256, 0, stream>>>(memory, pos, bn_gamma, bn_beta,
                                                         bn_mean, bn_var, conv_w, conv_b, ybf);
    k_w2bf<<<CDIM * CDIM / 256, 256, 0, stream>>>(featW, wbf);
    k_gemm_feat<<<dim3(BN / 16, 2), 128, 0, stream>>>(ybf, wbf, featb, mproj);
    k_rownorm<<<(int)BN, 256, 0, stream>>>(mproj, ybf);   // ybf now = normalized mem, bf16
    k_sim<<<dim3(NDIM / 128, HDIM, BDIM_B), 256, 0, stream>>>(ybf, tbf, mmask, flag, argb);
    k_gather<<<(int)BQ, 256, 0, stream>>>(mproj, tn, argb, g, gm);
    k_linear<<<(int)BQ, 256, 0, stream>>>(gm, projW, projb, pout);
    k_colnorm<<<BDIM_B, 256, 0, stream>>>(pout, nrm);
    k_finish<<<(int)BQ, 256, 0, stream>>>(pout, nrm, alpha, g, finW, finb, tgt, out);
}
